// Attention_90417651516253
// MI455X (gfx1250) — matrix operand order repaired and checked by reading, not compiled, not run on hardware
//
#include <hip/hip_runtime.h>
#include <cstdint>

// ---------------------------------------------------------------------------
// Problem constants (from reference)
// ---------------------------------------------------------------------------
#define BB   2
#define NN   2048
#define DIM  1024
#define HH   16
#define DH   64
#define INNER (HH * DH)          // 1024
#define EQKV  (3 * INNER)        // 3072
#define MROWS (BB * NN)          // 4096
#define PER_MAT ((size_t)BB * HH * NN * DH)   // 4,194,304 elements per q/k/v

typedef __attribute__((ext_vector_type(16))) __bf16 v16bf;
typedef __attribute__((ext_vector_type(8)))  float  v8f;
typedef __attribute__((ext_vector_type(4)))  unsigned int u32x4;
typedef __attribute__((ext_vector_type(8)))  int    i32x8;
typedef __attribute__((ext_vector_type(4)))  int    i32x4;

union FragU {
    uint4 u[2];
    v16bf v;
};
union Pack16 {
    uint4    u[2];
    uint16_t h[16];
};

// ---------------------------------------------------------------------------
// bf16 <-> f32 helpers (RNE)
// ---------------------------------------------------------------------------
__device__ __forceinline__ uint16_t f2bf(float f) {
    union { float f; uint32_t u; } v; v.f = f;
    uint32_t r = v.u + 0x7FFFu + ((v.u >> 16) & 1u);
    return (uint16_t)(r >> 16);
}
__device__ __forceinline__ float bf2f(uint16_t h) {
    union { uint32_t u; float f; } v; v.u = ((uint32_t)h) << 16;
    return v.f;
}
__device__ __forceinline__ uint32_t lds_addr32(const void* p) {
    // generic LDS pointer: low 32 bits are the workgroup-relative LDS byte addr
    return (uint32_t)(uintptr_t)p;
}

// ---------------------------------------------------------------------------
// Tensor Data Mover: 2D bf16 tile load, descriptor per ISA 08_async_tensor §8.
//  - tile_w elems/row, tile_h rows; global row stride in elements
//  - pad_interval/pad_amount codes: pad LDS rows (interval 2^(pi+1) DWORDs,
//    amount (pa+1) DWORDs). We use 16B pad (pa=3) per row.
// ---------------------------------------------------------------------------
__device__ __forceinline__ void tdm_load_2d_bf16(
        const uint16_t* gsrc, uint32_t lds_byte_addr,
        uint32_t tile_w, uint32_t tile_h, uint64_t row_stride,
        uint32_t pad_interval, uint32_t pad_amount)
{
    const uint64_t ga = (uint64_t)(uintptr_t)gsrc;
    u32x4 g0;
    g0[0] = 1u;                                            // count=1, user desc
    g0[1] = lds_byte_addr;                                 // lds_addr
    g0[2] = (uint32_t)ga;                                  // global_addr[31:0]
    g0[3] = ((uint32_t)(ga >> 32) & 0x01FFFFFFu)           // global_addr[56:32]
          | (2u << 30);                                    // type = 2 (image)
    // group1, assembled as 4 qwords (bit offsets per D# spec)
    const uint64_t q0 = ((uint64_t)1u << 16)               // data_size = 2B
                      | ((uint64_t)1u << 20)               // pad_enable
                      | ((uint64_t)pad_interval << 22)
                      | ((uint64_t)pad_amount << 25)
                      | ((uint64_t)(tile_w & 0xFFFFu) << 48);   // tensor_dim0 lo
    const uint64_t q1 = ((uint64_t)(tile_w >> 16))              // tensor_dim0 hi
                      | ((uint64_t)tile_h << 16)                // tensor_dim1
                      | ((uint64_t)(tile_w & 0xFFFFu) << 48);   // tile_dim0
    const uint64_t q2 = ((uint64_t)(tile_h & 0xFFFFu))          // tile_dim1
                      | ((row_stride & 0xFFFFFFFFull) << 32);   // dim0_stride lo
    const uint64_t q3 = (row_stride >> 32) & 0xFFFFull;         // dim0_stride hi
    i32x8 g1;
    g1[0] = (int)(uint32_t)q0; g1[1] = (int)(uint32_t)(q0 >> 32);
    g1[2] = (int)(uint32_t)q1; g1[3] = (int)(uint32_t)(q1 >> 32);
    g1[4] = (int)(uint32_t)q2; g1[5] = (int)(uint32_t)(q2 >> 32);
    g1[6] = (int)(uint32_t)q3; g1[7] = (int)(uint32_t)(q3 >> 32);
    const i32x4 z4 = {0, 0, 0, 0};
#if __clang_major__ >= 23
    const i32x8 z8 = {0, 0, 0, 0, 0, 0, 0, 0};
    __builtin_amdgcn_tensor_load_to_lds(g0, g1, z4, z4, z8, 0);
#else
    __builtin_amdgcn_tensor_load_to_lds(g0, g1, z4, z4, 0);
#endif
}
#define TDM_WAIT0() __builtin_amdgcn_s_wait_tensorcnt((short)0)

// ---------------------------------------------------------------------------
// WMMA fragment loaders (ISA 7.12.2 layouts, wave32)
// ---------------------------------------------------------------------------
__device__ __forceinline__ v16bf load_a16(const uint16_t* p0, int stride) {
    int lane = threadIdx.x & 31;
    const uint16_t* p = p0 + (size_t)(lane & 15) * stride + ((lane >> 4) << 3);
    FragU f;
    f.u[0] = *(const uint4*)(p);
    f.u[1] = *(const uint4*)(p + 16);
    return f.v;
}
__device__ __forceinline__ v16bf load_b16(const uint16_t* p0, int stride) {
    int lane = threadIdx.x & 31;
    const uint16_t* p = p0 + (size_t)(lane & 15) * stride + ((lane >> 4) << 3);
    FragU f;
    f.u[0] = *(const uint4*)(p);
    f.u[1] = *(const uint4*)(p + 16);
    return f.v;
}
#define WMMA_BF16(A, B, C) \
    __builtin_amdgcn_wmma_f32_16x16x32_bf16(false, (A), false, (B), (short)0, (C), false, false)

// ---------------------------------------------------------------------------
// Kernel 0: f32 -> bf16 convert
// ---------------------------------------------------------------------------
__global__ __launch_bounds__(256) void convert_bf16_kernel(
        const float* __restrict__ src, uint16_t* __restrict__ dst, int n) {
    int i = blockIdx.x * 256 + threadIdx.x;
    if (i < n) dst[i] = f2bf(src[i]);
}

// ---------------------------------------------------------------------------
// Shared GEMM core: C[128x128] += A[128xK] * B[128xK]^T with TDM-fed,
// double-buffered LDS (one barrier per K-step). 8 waves, wave tile 64x32.
// ---------------------------------------------------------------------------
#define G_LSTR 40   // padded LDS row stride (32 elems + 16B pad)

__device__ __forceinline__ void gemm_core(
        const uint16_t* __restrict__ Abase,   // rows at m0, stride lda
        const uint16_t* __restrict__ Bbase,   // rows at n0, stride lda
        int lda, int ksteps,
        uint16_t (*sA)[128 * G_LSTR], uint16_t (*sB)[128 * G_LSTR],
        int wave, int wm, int wn, v8f acc[4][2])
{
    if (wave == 0) {
        tdm_load_2d_bf16(Abase, lds_addr32(sA[0]), 32, 128, (uint64_t)lda, 3, 3);
        tdm_load_2d_bf16(Bbase, lds_addr32(sB[0]), 32, 128, (uint64_t)lda, 3, 3);
    }
    for (int kt = 0; kt < ksteps; ++kt) {
        const int cur = kt & 1;
        if (wave == 0) TDM_WAIT0();       // tile kt resident in LDS
        __syncthreads();
        if (wave == 0 && kt + 1 < ksteps) {   // prefetch tile kt+1 -> other buf
            const int k0 = (kt + 1) * 32;
            tdm_load_2d_bf16(Abase + k0, lds_addr32(sA[cur ^ 1]), 32, 128,
                             (uint64_t)lda, 3, 3);
            tdm_load_2d_bf16(Bbase + k0, lds_addr32(sB[cur ^ 1]), 32, 128,
                             (uint64_t)lda, 3, 3);
        }
        v16bf af[4], bf[2];
#pragma unroll
        for (int mi = 0; mi < 4; ++mi)
            af[mi] = load_a16(sA[cur] + (wm * 64 + mi * 16) * G_LSTR, G_LSTR);
#pragma unroll
        for (int ni = 0; ni < 2; ++ni)
            bf[ni] = load_b16(sB[cur] + (wn * 32 + ni * 16) * G_LSTR, G_LSTR);
#pragma unroll
        for (int mi = 0; mi < 4; ++mi)
#pragma unroll
            for (int ni = 0; ni < 2; ++ni)
                acc[mi][ni] = WMMA_BF16(af[mi], bf[ni], acc[mi][ni]);
    }
}

// ---------------------------------------------------------------------------
// Kernel 1: qkv = x @ w_qkv^T  (M=4096, N=3072, K=1024)
// Output scattered as bf16 into [3][B][H][N][DH].
// ---------------------------------------------------------------------------
__global__ __launch_bounds__(256) void gemm_qkv_kernel(
        const uint16_t* __restrict__ xb,     // [4096][1024]
        const uint16_t* __restrict__ wqkvb,  // [3072][1024]
        uint16_t* __restrict__ qkvb)         // [3][B][H][N][DH]
{
    __shared__ __align__(16) uint16_t sA[2][128 * G_LSTR];
    __shared__ __align__(16) uint16_t sB[2][128 * G_LSTR];

    const int tid  = threadIdx.x;
    const int wave = tid >> 5;
    const int wm   = wave >> 2;
    const int wn   = wave & 3;
    const int n0   = blockIdx.x * 128;
    const int m0   = blockIdx.y * 128;

    v8f acc[4][2] = {};
    gemm_core(xb + (size_t)m0 * DIM, wqkvb + (size_t)n0 * DIM, DIM, DIM / 32,
              sA, sB, wave, wm, wn, acc);

    const int lane  = tid & 31;
    const int cn    = lane & 15;
    const int rbase = (lane >> 4) * 8;
#pragma unroll
    for (int mi = 0; mi < 4; ++mi) {
#pragma unroll
        for (int ni = 0; ni < 2; ++ni) {
            const int e     = n0 + wn * 32 + ni * 16 + cn;   // 0..3071
            const int which = e >> 10;
            const int h     = (e >> 6) & 15;
            const int d     = e & 63;
#pragma unroll
            for (int r = 0; r < 8; ++r) {
                const int m = m0 + wm * 64 + mi * 16 + rbase + r;
                const int b = m >> 11;
                const int n = m & (NN - 1);
                const size_t idx =
                    ((((size_t)which * BB + b) * HH + h) * NN + n) * DH + d;
                qkvb[idx] = f2bf(acc[mi][ni][r]);
            }
        }
    }
}

// ---------------------------------------------------------------------------
// Kernel 2: RoPE on q,k (in place, bf16). Folds softmax scale 0.125 into q.
// ---------------------------------------------------------------------------
__global__ __launch_bounds__(256) void rope_kernel(
        uint16_t* __restrict__ qkvb, const float* __restrict__ rot) {
    size_t t = (size_t)blockIdx.x * 256 + threadIdx.x;
    const int d = (int)(t & 31); t >>= 5;
    const int n = (int)(t & (NN - 1)); t >>= 11;
    const int h = (int)(t & 15); t >>= 4;
    const int b = (int)(t & 1); t >>= 1;
    const int which = (int)t;   // 0 = q, 1 = k

    const size_t base = (((size_t)which * BB + b) * HH + h) * NN * DH + (size_t)n * DH;
    float lo = bf2f(qkvb[base + d]);
    float hi = bf2f(qkvb[base + d + 32]);
    const float p_lo = rot[(size_t)n * DH + d];
    const float p_hi = rot[(size_t)n * DH + d + 32];
    float o_lo = lo * __cosf(p_lo) - hi * __sinf(p_lo);
    float o_hi = hi * __cosf(p_hi) + lo * __sinf(p_hi);
    if (which == 0) { o_lo *= 0.125f; o_hi *= 0.125f; }   // DH^-0.5 exactly
    qkvb[base + d]      = f2bf(o_lo);
    qkvb[base + d + 32] = f2bf(o_hi);
}

// ---------------------------------------------------------------------------
// Kernel 3: flash attention per (b,h). 128 threads = 4 waves; each wave owns
// 16 query rows; K tiles staged by TDM, V transposed manually.
// ---------------------------------------------------------------------------
#define K_STR 72   // 64 elems + 16B pad (matches TDM pad_interval=4, amount=3)
#define V_STR 40
#define P_STR 40

__global__ __launch_bounds__(128) void attn_kernel(
        const uint16_t* __restrict__ qkvb,   // [3][B][H][N][DH] (q pre-scaled)
        uint16_t* __restrict__ attnout)      // [B][N][INNER] bf16
{
    __shared__ __align__(16) uint16_t sK[32 * K_STR];
    __shared__ __align__(16) uint16_t sV[DH * V_STR];       // transposed [d][key]
    __shared__ __align__(16) uint16_t sP[4][16 * P_STR];

    const int tid  = threadIdx.x;
    const int lane = tid & 31;
    const int wave = tid >> 5;

    const int blk  = blockIdx.x;          // B*H*(N/64)
    const int qblk = blk & 31;
    const int bh   = blk >> 5;
    const int b    = bh >> 4;
    const int h    = bh & 15;

    const size_t head_off = (((size_t)b * HH + h) * NN) * DH;
    const uint16_t* Qp = qkvb + head_off;
    const uint16_t* Kp = qkvb + PER_MAT + head_off;
    const uint16_t* Vp = qkvb + 2 * PER_MAT + head_off;

    const int q0 = qblk * 64 + wave * 16;

    v16bf qf[2];
#pragma unroll
    for (int kf = 0; kf < 2; ++kf)
        qf[kf] = load_a16(Qp + (size_t)q0 * DH + kf * 32, DH);

    float m_i[8], l_i[8];
#pragma unroll
    for (int r = 0; r < 8; ++r) { m_i[r] = -1.0e30f; l_i[r] = 0.0f; }
    v8f o_acc[4] = {};

    for (int j0 = 0; j0 < NN; j0 += 32) {
        __syncthreads();
        if (wave == 0)   // K tile [32 keys][64 dh] -> LDS rows padded to 72
            tdm_load_2d_bf16(Kp + (size_t)j0 * DH, lds_addr32(sK),
                             64, 32, DH, 4, 3);
        {   // V transposed: [d][key]
            const int key = tid >> 2, dgrp = tid & 3;
            Pack16 t;
            const uint4* src = (const uint4*)(Vp + (size_t)(j0 + key) * DH + dgrp * 16);
            t.u[0] = src[0]; t.u[1] = src[1];
#pragma unroll
            for (int i = 0; i < 16; ++i)
                sV[(dgrp * 16 + i) * V_STR + key] = t.h[i];
        }
        if (wave == 0) TDM_WAIT0();
        __syncthreads();

        // S = Q·K^T for two 16-key groups (scale folded into q)
        v8f s[2];
#pragma unroll
        for (int jt = 0; jt < 2; ++jt) {
            v8f t = {};
            t = WMMA_BF16(qf[0], load_b16(sK + jt * 16 * K_STR + 0,  K_STR), t);
            t = WMMA_BF16(qf[1], load_b16(sK + jt * 16 * K_STR + 32, K_STR), t);
            s[jt] = t;
        }

        // online softmax; rows live in 16-lane halves, reduce with shfl_xor
        float alpha[8];
#pragma unroll
        for (int r = 0; r < 8; ++r) {
            float mx = fmaxf(s[0][r], s[1][r]);
#pragma unroll
            for (int d = 1; d < 16; d <<= 1)
                mx = fmaxf(mx, __shfl_xor(mx, d, 32));
            const float m_new = fmaxf(m_i[r], mx);
            const float p0 = __expf(s[0][r] - m_new);
            const float p1 = __expf(s[1][r] - m_new);
            float ps = p0 + p1;
#pragma unroll
            for (int d = 1; d < 16; d <<= 1)
                ps += __shfl_xor(ps, d, 32);
            const float a = __expf(m_i[r] - m_new);
            l_i[r] = l_i[r] * a + ps;
            m_i[r] = m_new;
            alpha[r] = a;
            s[0][r] = p0;
            s[1][r] = p1;
        }

        // restage P (D layout -> A layout) through per-wave LDS patch
        uint16_t* myP = &sP[wave][0];
        {
            const int cn = lane & 15, rb = (lane >> 4) * 8;
#pragma unroll
            for (int jt = 0; jt < 2; ++jt)
#pragma unroll
                for (int r = 0; r < 8; ++r)
                    myP[(rb + r) * P_STR + jt * 16 + cn] = f2bf(s[jt][r]);
        }
        const v16bf pf = load_a16(myP, P_STR);

        // O = diag(alpha)·O + P·V
#pragma unroll
        for (int c = 0; c < 4; ++c) {
#pragma unroll
            for (int r = 0; r < 8; ++r) o_acc[c][r] *= alpha[r];
            o_acc[c] = WMMA_BF16(pf, load_b16(sV + c * 16 * V_STR, V_STR), o_acc[c]);
        }
    }

    const int cn = lane & 15, rb = (lane >> 4) * 8;
#pragma unroll
    for (int c = 0; c < 4; ++c) {
        const int e = h * DH + c * 16 + cn;
#pragma unroll
        for (int r = 0; r < 8; ++r) {
            const int row = q0 + rb + r;
            attnout[((size_t)b * NN + row) * INNER + e] = f2bf(o_acc[c][r] / l_i[r]);
        }
    }
}

// ---------------------------------------------------------------------------
// Kernel 4: out = attnout @ w_out^T + b_out (M=4096, N=1024, K=1024), f32 out
// ---------------------------------------------------------------------------
__global__ __launch_bounds__(256) void gemm_out_kernel(
        const uint16_t* __restrict__ ab,
        const uint16_t* __restrict__ wb,
        const float* __restrict__ bias,
        float* __restrict__ out)
{
    __shared__ __align__(16) uint16_t sA[2][128 * G_LSTR];
    __shared__ __align__(16) uint16_t sB[2][128 * G_LSTR];

    const int tid  = threadIdx.x;
    const int wave = tid >> 5;
    const int wm   = wave >> 2;
    const int wn   = wave & 3;
    const int n0   = blockIdx.x * 128;
    const int m0   = blockIdx.y * 128;

    v8f acc[4][2] = {};
    gemm_core(ab + (size_t)m0 * INNER, wb + (size_t)n0 * INNER, INNER, INNER / 32,
              sA, sB, wave, wm, wn, acc);

    const int lane = tid & 31, cn = lane & 15, rbase = (lane >> 4) * 8;
#pragma unroll
    for (int mi = 0; mi < 4; ++mi) {
#pragma unroll
        for (int ni = 0; ni < 2; ++ni) {
            const int dcol = n0 + wn * 32 + ni * 16 + cn;
            const float bv = bias[dcol];
#pragma unroll
            for (int r = 0; r < 8; ++r) {
                const int m = m0 + wm * 64 + mi * 16 + rbase + r;
                out[(size_t)m * DIM + dcol] = acc[mi][ni][r] + bv;
            }
        }
    }
}

// ---------------------------------------------------------------------------
// Launcher
// ---------------------------------------------------------------------------
extern "C" void kernel_launch(void* const* d_in, const int* in_sizes, int n_in,
                              void* d_out, int out_size, void* d_ws, size_t ws_size,
                              hipStream_t stream) {
    const float* x      = (const float*)d_in[0];
    const float* rot    = (const float*)d_in[1];
    const float* w_qkv  = (const float*)d_in[2];
    const float* w_out  = (const float*)d_in[3];
    const float* b_out  = (const float*)d_in[4];
    float*       out    = (float*)d_out;

    uint16_t* ws     = (uint16_t*)d_ws;
    uint16_t* xb     = ws;                                  // 4096*1024
    uint16_t* wqkvb  = xb    + (size_t)MROWS * DIM;         // 3072*1024
    uint16_t* woutb  = wqkvb + (size_t)EQKV * DIM;          // 1024*1024
    uint16_t* qkvb   = woutb + (size_t)DIM * INNER;         // 3*PER_MAT
    uint16_t* attnout = xb;                                 // xb dead after gemm1

    convert_bf16_kernel<<<(MROWS * DIM) / 256, 256, 0, stream>>>(x, xb, MROWS * DIM);
    convert_bf16_kernel<<<(EQKV * DIM) / 256, 256, 0, stream>>>(w_qkv, wqkvb, EQKV * DIM);
    convert_bf16_kernel<<<(DIM * INNER) / 256, 256, 0, stream>>>(w_out, woutb, DIM * INNER);

    gemm_qkv_kernel<<<dim3(EQKV / 128, MROWS / 128), 256, 0, stream>>>(xb, wqkvb, qkvb);

    rope_kernel<<<(unsigned)(2 * PER_MAT / 64 * 32 / 256), 256, 0, stream>>>(qkvb, rot);

    attn_kernel<<<BB * HH * (NN / 64), 128, 0, stream>>>(qkvb, attnout);

    gemm_out_kernel<<<dim3(DIM / 128, MROWS / 128), 256, 0, stream>>>(attnout, woutb, b_out, out);
}
